// HierarchicalClassificationHead_17179869184278
// MI455X (gfx1250) — matrix operand order repaired and checked by reading, not compiled, not run on hardware
//
#include <hip/hip_runtime.h>

typedef __attribute__((ext_vector_type(16))) __bf16 v16bf;
typedef __attribute__((ext_vector_type(8)))  float  v8f;

#define TILE_M 128
#define TILE_N 128
#define BK 32
#define LDS_STRIDE 40   // halfwords per row: 32 data + 8 pad (80B row, 16B aligned)

union FragBF { uint4 u[2]; v16bf v; };

__device__ __forceinline__ unsigned short f2bf(float f) {
    unsigned int u = __float_as_uint(f);
    unsigned int r = u + 0x7FFFu + ((u >> 16) & 1u);   // round-to-nearest-even
    return (unsigned short)(r >> 16);
}

// CDNA5 async global->LDS DMA (16B per lane), tracked by ASYNCcnt
__device__ __forceinline__ void async_ld16(unsigned ldsAddr, unsigned long long gAddr) {
    asm volatile("global_load_async_to_lds_b128 %0, %1, off"
                 :: "v"(ldsAddr), "v"(gAddr) : "memory");
}
template <int N>
__device__ __forceinline__ void wait_asynccnt() {
    asm volatile("s_wait_asynccnt %0" :: "n"(N) : "memory");
}
__device__ __forceinline__ unsigned lds_addr32(const void* p) {
    // flat LDS aperture keeps the LDS byte offset in addr[31:0]
    return (unsigned)(unsigned long long)p;
}

// ---------------- f32 -> bf16 conversion (x4 vectorized) ----------------
__global__ __launch_bounds__(256) void cvt_f32_bf16_x4(
    const float* __restrict__ s, unsigned short* __restrict__ d, int n4)
{
    int i = blockIdx.x * 256 + threadIdx.x;
    if (i < n4) {
        float4 f = ((const float4*)s)[i];
        ushort4 h;
        h.x = f2bf(f.x); h.y = f2bf(f.y); h.z = f2bf(f.z); h.w = f2bf(f.w);
        ((ushort4*)d)[i] = h;
    }
}

// ---------------- bf16 WMMA GEMM: Z = sigmoid(x @ W^T + b) ----------------
// A: [4096,1024] bf16 row-major, Bw: [N,1024] bf16 row-major, Z: [4096,N] f32
__global__ __launch_bounds__(256) void gemm_bias_sigmoid(
    const unsigned short* __restrict__ A,
    const unsigned short* __restrict__ Bw,
    const float* __restrict__ bias,
    float* __restrict__ Z, int N)
{
    __shared__ unsigned short As[2][TILE_M * LDS_STRIDE];
    __shared__ unsigned short Bs[2][TILE_N * LDS_STRIDE];
    __shared__ unsigned short dump[8];   // sink for masked-off async B loads

    const int tid   = threadIdx.x;
    const int lane  = tid & 31;
    const int wave  = tid >> 5;
    const int waveM = wave & 3;    // 4 waves along M -> 32 rows each
    const int waveN = wave >> 2;   // 2 waves along N -> 64 cols each
    const int m0 = blockIdx.y * TILE_M;
    const int n0 = blockIdx.x * TILE_N;
    const int half = lane >> 4;    // 0: lanes 0-15, 1: lanes 16-31
    const int l16  = lane & 15;

    // per-thread staging assignment: 2 chunks of 8 halfwords for A, 2 for B
    const int r0   = tid >> 2;             // chunk 0 row
    const int r1   = (tid + 256) >> 2;     // chunk 1 row
    const int col0 = (tid & 3) << 3;
    const int nr0  = n0 + r0, nr1 = n0 + r1;
    const int nc0  = (nr0 < N) ? nr0 : (N - 1);   // clamp to a valid global row
    const int nc1  = (nr1 < N) ? nr1 : (N - 1);
    const unsigned dumpAddr = lds_addr32(&dump[0]);

    // prologue: zero LDS rows of B that will never receive DMA (ragged N tail)
    if (n0 + TILE_N > N) {
        const uint4 zz = make_uint4(0u, 0u, 0u, 0u);
        for (int c = tid; c < TILE_N * 4; c += 256) {
            int r = c >> 2, col = (c & 3) << 3;
            if (n0 + r >= N) {
                *(uint4*)&Bs[0][r * LDS_STRIDE + col] = zz;
                *(uint4*)&Bs[1][r * LDS_STRIDE + col] = zz;
            }
        }
    }

    v8f acc[2][4];
#pragma unroll
    for (int i = 0; i < 2; ++i)
#pragma unroll
        for (int j = 0; j < 4; ++j)
            acc[i][j] = (v8f)(0.0f);

    // always issues exactly 4 async loads per thread (constant ASYNCcnt delta)
    auto issue_tile = [&](int k0, int buf) {
        async_ld16(lds_addr32(&As[buf][r0 * LDS_STRIDE + col0]),
                   (unsigned long long)(const void*)(A + (((size_t)(m0 + r0)) << 10) + k0 + col0));
        async_ld16(lds_addr32(&As[buf][r1 * LDS_STRIDE + col0]),
                   (unsigned long long)(const void*)(A + (((size_t)(m0 + r1)) << 10) + k0 + col0));
        async_ld16((nr0 < N) ? lds_addr32(&Bs[buf][r0 * LDS_STRIDE + col0]) : dumpAddr,
                   (unsigned long long)(const void*)(Bw + (((size_t)nc0) << 10) + k0 + col0));
        async_ld16((nr1 < N) ? lds_addr32(&Bs[buf][r1 * LDS_STRIDE + col0]) : dumpAddr,
                   (unsigned long long)(const void*)(Bw + (((size_t)nc1) << 10) + k0 + col0));
    };

    issue_tile(0, 0);

    int buf = 0;
    for (int k0 = 0; k0 < 1024; k0 += BK) {
        if (k0 + BK < 1024) {
            issue_tile(k0 + BK, buf ^ 1);   // DMA next tile while computing current
            wait_asynccnt<4>();             // in-order: current tile's 4 loads done
        } else {
            wait_asynccnt<0>();
        }
        __syncthreads();                    // all waves' DMA for this tile landed

        const unsigned short* Asb = As[buf];
        const unsigned short* Bsb = Bs[buf];

        // A fragment (16-bit A 16x32): lane<16 K=0..7,16..23; lane>=16 K=8..15,24..31
        FragBF a[2], b[4];
#pragma unroll
        for (int i = 0; i < 2; ++i) {
            int row = waveM * 32 + i * 16 + l16;
            const unsigned short* p = &Asb[row * LDS_STRIDE + half * 8];
            a[i].u[0] = *(const uint4*)p;          // K = half*8 .. +7
            a[i].u[1] = *(const uint4*)(p + 16);   // K = 16+half*8 .. +7
        }
        // B fragment (16-bit B 32x16): lane<16 N=lane,K=0..7,16..23; lane>=16 K=8..15,24..31
#pragma unroll
        for (int j = 0; j < 4; ++j) {
            int row = waveN * 64 + j * 16 + l16;
            const unsigned short* p = &Bsb[row * LDS_STRIDE + half * 8];
            b[j].u[0] = *(const uint4*)p;          // K = half*8 .. +7
            b[j].u[1] = *(const uint4*)(p + 16);   // K = 16+half*8 .. +7
        }
#pragma unroll
        for (int i = 0; i < 2; ++i)
#pragma unroll
            for (int j = 0; j < 4; ++j)
                acc[i][j] = __builtin_amdgcn_wmma_f32_16x16x32_bf16(
                    false, a[i].v, false, b[j].v, (short)0, acc[i][j], false, false);

        __syncthreads();                    // safe to overwrite this buffer next iter
        buf ^= 1;
    }

    // epilogue: C layout -> vgpr r holds M=r (lanes 0-15) / M=r+8 (lanes 16-31), N=lane%16
#pragma unroll
    for (int i = 0; i < 2; ++i) {
        int mloc = waveM * 32 + i * 16 + half * 8;
#pragma unroll
        for (int j = 0; j < 4; ++j) {
            int n = n0 + waveN * 64 + j * 16 + l16;
            if (n < N) {
                float bv = bias[n];
#pragma unroll
                for (int r = 0; r < 8; ++r) {
                    float v = acc[i][j][r] + bv;
                    Z[(size_t)(m0 + mloc + r) * N + n] = 1.0f / (1.0f + __expf(-v));
                }
            }
        }
    }
}

// ---------------- path products + per-row cross-entropy ----------------
// depth 1: l=z[k]; depth 2: l=z[k/10]*z[k]; depth 3: l=z[k/100]*z[k/10]*z[k] (in-place -> logits)
__global__ __launch_bounds__(256) void path_ce(
    float* __restrict__ Z, const int* __restrict__ labels,
    float* __restrict__ lossRows, int n, int depth, int labDiv)
{
    __shared__ float head[280];
    __shared__ float red[256];
    const int b   = blockIdx.x;
    const int tid = threadIdx.x;
    float* row = Z + (size_t)b * n;
    const int hn = n < 280 ? n : 280;
    for (int i = tid; i < hn; i += 256) head[i] = row[i];
    __syncthreads();

    float lmax = -3.4e38f;
    for (int k = tid; k < n; k += 256) {
        float l;
        if (depth == 1)      l = head[k];
        else if (depth == 2) l = head[k / 10] * head[k];
        else {
            float zk = (k < 280) ? head[k] : row[k];
            l = head[k / 100] * head[k / 10] * zk;
            row[k] = l;   // write logits3 into d_out
        }
        lmax = fmaxf(lmax, l);
    }
    red[tid] = lmax; __syncthreads();
    for (int s = 128; s > 0; s >>= 1) {
        if (tid < s) red[tid] = fmaxf(red[tid], red[tid + s]);
        __syncthreads();
    }
    const float maxv = red[0]; __syncthreads();

    float lsum = 0.0f;
    for (int k = tid; k < n; k += 256) {
        float l;
        if (depth == 1)      l = head[k];
        else if (depth == 2) l = head[k / 10] * head[k];
        else                 l = row[k];
        lsum += __expf(l - maxv);
    }
    red[tid] = lsum; __syncthreads();
    for (int s = 128; s > 0; s >>= 1) {
        if (tid < s) red[tid] += red[tid + s];
        __syncthreads();
    }
    if (tid == 0) {
        int lab = labels[b] / labDiv;
        float ll;
        if (depth == 1)      ll = head[lab];
        else if (depth == 2) ll = head[lab / 10] * head[lab];
        else                 ll = row[lab];
        lossRows[b] = maxv + __logf(red[0]) - ll;
    }
}

// ---------------- final loss reduction ----------------
__global__ __launch_bounds__(256) void reduce_loss(
    const float* __restrict__ lossRows, float* __restrict__ out, int count, float scale)
{
    __shared__ float red[256];
    float s = 0.0f;
    for (int i = threadIdx.x; i < count; i += 256) s += lossRows[i];
    red[threadIdx.x] = s; __syncthreads();
    for (int st = 128; st > 0; st >>= 1) {
        if (threadIdx.x < st) red[threadIdx.x] += red[threadIdx.x + st];
        __syncthreads();
    }
    if (threadIdx.x == 0) out[0] = red[0] * scale;
}

extern "C" void kernel_launch(void* const* d_in, const int* in_sizes, int n_in,
                              void* d_out, int out_size, void* d_ws, size_t ws_size,
                              hipStream_t stream) {
    (void)in_sizes; (void)n_in; (void)out_size; (void)ws_size;
    const float* x      = (const float*)d_in[0];
    const int*   labels = (const int*)d_in[1];
    const float* W1 = (const float*)d_in[2]; const float* b1 = (const float*)d_in[3];
    const float* W2 = (const float*)d_in[4]; const float* b2 = (const float*)d_in[5];
    const float* W3 = (const float*)d_in[6]; const float* b3 = (const float*)d_in[7];
    float* out = (float*)d_out;

    const size_t Bsz = 4096, H = 1024;
    const int N1 = 28, N2 = 280, N3 = 2800;

    char* ws = (char*)d_ws;
    size_t off = 0;
    unsigned short* xb  = (unsigned short*)(ws + off); off += Bsz * H * 2;        // 8 MB
    unsigned short* w1b = (unsigned short*)(ws + off); off += (size_t)N1 * H * 2;
    unsigned short* w2b = (unsigned short*)(ws + off); off += (size_t)N2 * H * 2;
    unsigned short* w3b = (unsigned short*)(ws + off); off += (size_t)N3 * H * 2;
    float* z1    = (float*)(ws + off); off += Bsz * N1 * 4;
    float* z2    = (float*)(ws + off); off += Bsz * N2 * 4;
    float* loss1 = (float*)(ws + off); off += Bsz * 4;   // loss1..loss3 contiguous
    float* loss2 = (float*)(ws + off); off += Bsz * 4;
    float* loss3 = (float*)(ws + off); off += Bsz * 4;
    (void)loss2; (void)loss3;

    // 1) bf16 conversion
    cvt_f32_bf16_x4<<<(int)(Bsz * H / 4 / 256), 256, 0, stream>>>(x, xb, (int)(Bsz * H / 4));
    cvt_f32_bf16_x4<<<(N1 * 1024 / 4 + 255) / 256, 256, 0, stream>>>(W1, w1b, N1 * 1024 / 4);
    cvt_f32_bf16_x4<<<(N2 * 1024 / 4 + 255) / 256, 256, 0, stream>>>(W2, w2b, N2 * 1024 / 4);
    cvt_f32_bf16_x4<<<(N3 * 1024 / 4 + 255) / 256, 256, 0, stream>>>(W3, w3b, N3 * 1024 / 4);

    // 2) z_l = sigmoid(x @ Wl^T + bl); level 3 writes straight into d_out
    gemm_bias_sigmoid<<<dim3((N1 + TILE_N - 1) / TILE_N, (int)(Bsz / TILE_M)), 256, 0, stream>>>(
        xb, w1b, b1, z1, N1);
    gemm_bias_sigmoid<<<dim3((N2 + TILE_N - 1) / TILE_N, (int)(Bsz / TILE_M)), 256, 0, stream>>>(
        xb, w2b, b2, z2, N2);
    gemm_bias_sigmoid<<<dim3((N3 + TILE_N - 1) / TILE_N, (int)(Bsz / TILE_M)), 256, 0, stream>>>(
        xb, w3b, b3, out, N3);

    // 3) path products (logits3 in place in d_out) + per-row CE
    path_ce<<<(int)Bsz, 256, 0, stream>>>(z1, labels, loss1, N1, 1, 100);
    path_ce<<<(int)Bsz, 256, 0, stream>>>(z2, labels, loss2, N2, 2, 10);
    path_ce<<<(int)Bsz, 256, 0, stream>>>(out, labels, loss3, N3, 3, 1);

    // 4) loss = (sum of all per-row NLLs) / B  -> d_out[B*N3]
    reduce_loss<<<1, 256, 0, stream>>>(loss1, out + Bsz * (size_t)N3, (int)(3 * Bsz), 1.0f / 4096.0f);
}
